// MultiheadAttention_48962627174889
// MI455X (gfx1250) — matrix operand order repaired and checked by reading, not compiled, not run on hardware
//
#include <hip/hip_runtime.h>

// ---------------------------------------------------------------------------
// MHA forward for MI455X (gfx1250, wave32, WMMA).
// All three GEMM-shaped stages run on v_wmma_f32_16x16x32_bf16 with f32 accum.
// Output-projection A-tiles are staged LDS-side by the Tensor Data Mover.
// ---------------------------------------------------------------------------

#define B_  2
#define S_  2048
#define D_  1024
#define H_  16
#define HD_ 64
#define M_  (B_ * S_)     // 4096 rows
#define N3_ (3 * D_)      // 3072 qkv output cols

typedef __attribute__((ext_vector_type(16))) __bf16 v16bf;
typedef __attribute__((ext_vector_type(8)))  float  v8f;
typedef unsigned int v4u __attribute__((ext_vector_type(4)));
typedef int          v8i __attribute__((ext_vector_type(8)));
typedef int          v4i __attribute__((ext_vector_type(4)));

union FragBF { v16bf v; uint4 q[2]; __bf16 e[16]; };
union AccF   { v8f   v; float f[8]; };

#if __has_builtin(__builtin_amdgcn_tensor_load_to_lds) && \
    __has_builtin(__builtin_amdgcn_s_wait_tensorcnt)
#define USE_TDM 1
#else
#define USE_TDM 0
#endif

__device__ __forceinline__ v8f wmma_bf16(v16bf a, v16bf b, v8f c) {
  // (neg_a, A, neg_b, B, c_mod, C, reuse_a, reuse_b)
  return __builtin_amdgcn_wmma_f32_16x16x32_bf16(false, a, false, b, (short)0, c,
                                                 false, false);
}

// GEMM tiling: block = 256 threads = 8 waves; tile 128(M) x 64(N), K-step 32.
// Wave (wm in 0..3, wn in 0..1) owns a 32x32 sub-tile = 2x2 WMMA tiles.
#define BM  128
#define BN  64
#define BK  32
#define LDT 40   // LDS row stride in bf16 (80B = 20 dwords; 16B-aligned rows)

// ---------------------------------------------------------------------------
// Kernel 1: fused QKV projection.  C[m,n] = query[m,:] . qkv_w[n,:] + qkv_b[n]
// Outputs bf16:  Q,K -> [B*H][S][64]  ;  V -> transposed [B*H][64][S]
// ---------------------------------------------------------------------------
__global__ __launch_bounds__(256)
void qkv_gemm_kernel(const float* __restrict__ Aq, const float* __restrict__ W,
                     const float* __restrict__ bias,
                     __bf16* __restrict__ Qb, __bf16* __restrict__ Kb,
                     __bf16* __restrict__ Vt) {
  __shared__ __bf16 sA[BM * LDT];
  __shared__ __bf16 sB[BN * LDT];
  const int mt = blockIdx.x & 31, nt = blockIdx.x >> 5;   // 32 x 48 tiles
  const int m0 = mt * BM, n0 = nt * BN;
  const int tid = threadIdx.x, lane = tid & 31, w = tid >> 5;
  const int wm = w & 3, wn = w >> 2;
  const int hsel = (lane >> 4) & 1, ln15 = lane & 15;

  AccF acc[2][2];
  #pragma unroll
  for (int i = 0; i < 2; i++)
    #pragma unroll
    for (int j = 0; j < 2; j++)
      #pragma unroll
      for (int r = 0; r < 8; r++) acc[i][j].f[r] = 0.0f;

  for (int k0 = 0; k0 < D_; k0 += BK) {
    { // stage A tile 128x32 fp32 -> bf16 LDS (16 elems / thread)
      const int row = tid >> 1, ch = (tid & 1) * 16;
      const float4* gp = (const float4*)(Aq + (size_t)(m0 + row) * D_ + k0 + ch);
      float4 f0 = gp[0], f1 = gp[1], f2 = gp[2], f3 = gp[3];
      float t[16] = {f0.x, f0.y, f0.z, f0.w, f1.x, f1.y, f1.z, f1.w,
                     f2.x, f2.y, f2.z, f2.w, f3.x, f3.y, f3.z, f3.w};
      union { __bf16 e[16]; uint4 q[2]; } cv;
      #pragma unroll
      for (int i = 0; i < 16; i++) cv.e[i] = (__bf16)t[i];
      uint4* lp = (uint4*)&sA[row * LDT + ch];
      lp[0] = cv.q[0]; lp[1] = cv.q[1];
      if (k0 + BK < D_)
        __builtin_prefetch(Aq + (size_t)(m0 + row) * D_ + k0 + BK + ch, 0, 1);
    }
    { // stage W tile 64x32 fp32 -> bf16 LDS (8 elems / thread)
      const int row = tid >> 2, ch = (tid & 3) * 8;
      const float4* gp = (const float4*)(W + (size_t)(n0 + row) * D_ + k0 + ch);
      float4 f0 = gp[0], f1 = gp[1];
      float t[8] = {f0.x, f0.y, f0.z, f0.w, f1.x, f1.y, f1.z, f1.w};
      union { __bf16 e[8]; uint4 q; } cv;
      #pragma unroll
      for (int i = 0; i < 8; i++) cv.e[i] = (__bf16)t[i];
      *(uint4*)&sB[row * LDT + ch] = cv.q;
      if (k0 + BK < D_)
        __builtin_prefetch(W + (size_t)(n0 + row) * D_ + k0 + BK + ch, 0, 1);
    }
    __syncthreads();

    // A fragments: 16x32, lane<16 holds K 0-15, lane>=16 K 16-31
    FragBF a[2], bfr[2];
    #pragma unroll
    for (int i = 0; i < 2; i++) {
      const int r = wm * 32 + i * 16 + ln15;
      a[i].q[0] = *(const uint4*)&sA[r * LDT + hsel * 16];
      a[i].q[1] = *(const uint4*)&sA[r * LDT + hsel * 16 + 8];
    }
    // B fragments: 32x16, lane = column, 16 contiguous K values
    #pragma unroll
    for (int j = 0; j < 2; j++) {
      const int c = wn * 32 + j * 16 + ln15;
      const uint4* p = (const uint4*)&sB[c * LDT + hsel * 16];
      bfr[j].q[0] = p[0]; bfr[j].q[1] = p[1];
    }
    #pragma unroll
    for (int i = 0; i < 2; i++)
      #pragma unroll
      for (int j = 0; j < 2; j++)
        acc[i][j].v = wmma_bf16(a[i].v, bfr[j].v, acc[i][j].v);
    __syncthreads();
  }

  // scatter: bias add, convert bf16, route q/k/v into attention layouts
  #pragma unroll
  for (int j = 0; j < 2; j++) {
    const int n = n0 + wn * 32 + j * 16 + ln15;
    const float bval = bias[n];
    const int part = n >> 10, loc = n & 1023, h = loc >> 6, d = loc & 63;
    #pragma unroll
    for (int i = 0; i < 2; i++) {
      const int mbase = m0 + wm * 32 + i * 16 + (hsel << 3);
      #pragma unroll
      for (int r = 0; r < 8; r++) {
        const int m = mbase + r;
        const int b = m >> 11, s = m & 2047;
        const int bh = (b << 4) + h;
        const __bf16 bv = (__bf16)(acc[i][j].f[r] + bval);
        if (part == 0)      Qb[((size_t)bh * S_ + s) * HD_ + d] = bv;
        else if (part == 1) Kb[((size_t)bh * S_ + s) * HD_ + d] = bv;
        else                Vt[((size_t)bh * HD_ + d) * S_ + s] = bv;
      }
    }
  }
}

// ---------------------------------------------------------------------------
// Kernel 2: flash attention, one wave per 16 query rows, key chunks of 32.
// Causal + key-padding mask, online softmax, f32 accumulators.
// Output Xb bf16 in merged-heads layout [B][S][H*64].
// ---------------------------------------------------------------------------
__global__ __launch_bounds__(256)
void attn_kernel(const __bf16* __restrict__ Qb, const __bf16* __restrict__ Kb,
                 const __bf16* __restrict__ Vt,
                 const unsigned char* __restrict__ pmask,
                 __bf16* __restrict__ Xb) {
  __shared__ __bf16 pbuf[8][16][32];   // per-wave P transpose scratch
  const int tid = threadIdx.x, lane = tid & 31, w = tid >> 5;
  const int job = blockIdx.x * 8 + w;        // 4096 jobs = 32 bh * 128 q-tiles
  const int q0 = (job & 127) << 4;
  const int bh = job >> 7;
  const int b = bh >> 4, h = bh & 15;
  const int hsel = (lane >> 4) & 1, ln15 = lane & 15;

  const __bf16* Qp = Qb + (size_t)bh * S_ * HD_;
  const __bf16* Kp = Kb + (size_t)bh * S_ * HD_;
  const __bf16* Vp = Vt + (size_t)bh * HD_ * S_;
  const unsigned char* pm = pmask + (size_t)b * S_;

  // Q fragments: two 16x32 A-frags covering d = 0..63
  FragBF aq[2];
  #pragma unroll
  for (int i = 0; i < 2; i++) {
    const __bf16* base = Qp + (size_t)(q0 + ln15) * HD_ + i * 32 + hsel * 16;
    aq[i].q[0] = *(const uint4*)(base);
    aq[i].q[1] = *(const uint4*)(base + 8);
  }

  AccF o[4];
  #pragma unroll
  for (int j = 0; j < 4; j++)
    #pragma unroll
    for (int r = 0; r < 8; r++) o[j].f[r] = 0.0f;
  float mrow[8], lrow[8];
  #pragma unroll
  for (int r = 0; r < 8; r++) { mrow[r] = -1e30f; lrow[r] = 0.0f; }

  const int qmax = q0 + 15;
  for (int k0 = 0; k0 <= qmax; k0 += 32) {
    // ---- scores: S[16q x 32k] as two 16x16 WMMA accumulators ----
    float sv[2][8];
    #pragma unroll
    for (int t = 0; t < 2; t++) {
      const int kt = k0 + t * 16;
      AccF s;
      #pragma unroll
      for (int r = 0; r < 8; r++) s.f[r] = 0.0f;
      #pragma unroll
      for (int dch = 0; dch < 2; dch++) {
        FragBF bk;   // B-frag: column = key, 16 contiguous d values
        const __bf16* kp = Kp + (size_t)(kt + ln15) * HD_ + dch * 32 + hsel * 16;
        bk.q[0] = *(const uint4*)(kp);
        bk.q[1] = *(const uint4*)(kp + 8);
        s.v = wmma_bf16(aq[dch].v, bk.v, s.v);
      }
      const int key = kt + ln15;
      const bool pad = pm[key] != 0;
      #pragma unroll
      for (int r = 0; r < 8; r++) {
        const int q = q0 + r + (hsel << 3);
        const float vv = s.f[r] * 0.125f;          // 1/sqrt(64)
        sv[t][r] = (pad || (key > q)) ? -1e30f : vv;
      }
    }
    if (k0 + 32 <= qmax) __builtin_prefetch(Kp + (size_t)(k0 + 32) * HD_, 0, 1);

    // ---- online softmax (row reductions across 16-lane halves) ----
    float rmax[8];
    #pragma unroll
    for (int r = 0; r < 8; r++) rmax[r] = fmaxf(sv[0][r], sv[1][r]);
    #pragma unroll
    for (int msk = 1; msk < 16; msk <<= 1)
      #pragma unroll
      for (int r = 0; r < 8; r++)
        rmax[r] = fmaxf(rmax[r], __shfl_xor(rmax[r], msk, 32));

    float psum[8];
    #pragma unroll
    for (int r = 0; r < 8; r++) {
      const float mn   = fmaxf(mrow[r], rmax[r]);
      const float corr = __expf(mrow[r] - mn);
      const float p0   = __expf(sv[0][r] - mn);
      const float p1   = __expf(sv[1][r] - mn);
      mrow[r] = mn;
      psum[r] = p0 + p1;
      lrow[r] *= corr;
      const int row = r + (hsel << 3);
      pbuf[w][row][ln15]      = (__bf16)p0;   // C-layout -> LDS
      pbuf[w][row][ln15 + 16] = (__bf16)p1;
      #pragma unroll
      for (int j = 0; j < 4; j++) o[j].f[r] *= corr;
    }
    #pragma unroll
    for (int msk = 1; msk < 16; msk <<= 1)
      #pragma unroll
      for (int r = 0; r < 8; r++) psum[r] += __shfl_xor(psum[r], msk, 32);
    #pragma unroll
    for (int r = 0; r < 8; r++) lrow[r] += psum[r];

    // ---- reload P as 16x32 A-frag, accumulate O += P @ V ----
    FragBF pf;
    {
      const __bf16* pb = &pbuf[w][ln15][hsel * 16];
      pf.q[0] = *(const uint4*)(pb);
      pf.q[1] = *(const uint4*)(pb + 8);
    }
    #pragma unroll
    for (int j = 0; j < 4; j++) {
      FragBF vf;   // B-frag from transposed V: column = d, contiguous keys
      const __bf16* vb = Vp + (size_t)(j * 16 + ln15) * S_ + k0 + hsel * 16;
      vf.q[0] = *(const uint4*)(vb);
      vf.q[1] = *(const uint4*)(vb + 8);
      o[j].v = wmma_bf16(pf.v, vf.v, o[j].v);
    }
  }

  // ---- finalize rows and store merged-heads bf16 ----
  #pragma unroll
  for (int r = 0; r < 8; r++) {
    const float inv = lrow[r] > 0.0f ? 1.0f / lrow[r] : 0.0f;
    const int q = q0 + r + (hsel << 3);
    const size_t base = (((size_t)b * S_ + q) * H_ + h) * HD_;
    #pragma unroll
    for (int j = 0; j < 4; j++)
      Xb[base + j * 16 + ln15] = (__bf16)(o[j].f[r] * inv);
  }
}

// ---------------------------------------------------------------------------
// Kernel 3: output projection.  out[m,n] = Xb[m,:] . out_w[n,:] + out_b[n]
// A-tile (bf16 128x32, row stride 2KB) is staged into LDS by the TDM with
// hardware padding (16 dwords data + 4 dwords pad = LDT row stride).
// ---------------------------------------------------------------------------
__global__ __launch_bounds__(256)
void out_gemm_kernel(const __bf16* __restrict__ Xb, const float* __restrict__ W,
                     const float* __restrict__ bias, float* __restrict__ Out) {
  __shared__ __bf16 sA[BM * LDT];
  __shared__ __bf16 sB[BN * LDT];
  const int mt = blockIdx.x & 31, nt = blockIdx.x >> 5;   // 32 x 16 tiles
  const int m0 = mt * BM, n0 = nt * BN;
  const int tid = threadIdx.x, lane = tid & 31, w = tid >> 5;
  const int wm = w & 3, wn = w >> 2;
  const int hsel = (lane >> 4) & 1, ln15 = lane & 15;

  AccF acc[2][2];
  #pragma unroll
  for (int i = 0; i < 2; i++)
    #pragma unroll
    for (int j = 0; j < 2; j++)
      #pragma unroll
      for (int r = 0; r < 8; r++) acc[i][j].f[r] = 0.0f;

  for (int k0 = 0; k0 < D_; k0 += BK) {
#if USE_TDM
    if (w == 0) {
      // Tensor DMA: 2D tile, 2B elems, tile_dim = [32, 128],
      // tensor_dim0_stride = 1024 elems; LDS pad: every 16 dwords add 4 dwords.
      const unsigned lds_base = (unsigned)(size_t)(&sA[0]);
      const unsigned long long ga =
          (unsigned long long)(size_t)(Xb + (size_t)m0 * D_ + k0);
      v4u g0 = { 1u,                                   // D# count = 1 (user)
                 lds_base,                             // lds_addr
                 (unsigned)(ga & 0xffffffffu),         // global_addr[31:0]
                 (unsigned)((ga >> 32) & 0x01ffffffu)  // global_addr[56:32]
                     | 0x80000000u };                  // type = 2 ("image")
      v8i g1 = { (int)((1u << 16)      // data_size = 1 (2 bytes)
                     | (1u << 20)      // pad_enable
                     | (3u << 22)      // pad_interval: 16 dwords
                     | (3u << 25)),    // pad_amount:   4 dwords
                 (int)(1024u << 16),   // tensor_dim0 = 1024 (bits [31:16])
                 (int)(4096u << 16),   // tensor_dim1 = 4096 (bits [31:16])
                 (int)(32u << 16),     // tile_dim0 = 32   (bits [31:16])
                 (int)128u,            // tile_dim1 = 128  (bits [15:0])
                 (int)1024u,           // tensor_dim0_stride[31:0] = 1024
                 0, 0 };
      v4i gz4 = { 0, 0, 0, 0 };                        // groups 2/3 disabled (2D)
      v8i gz8 = { 0, 0, 0, 0, 0, 0, 0, 0 };            // group 4 unused
      __builtin_amdgcn_tensor_load_to_lds(g0, g1, gz4, gz4, gz8, 0);
      __builtin_amdgcn_s_wait_tensorcnt((short)0);  // LDS valid before barrier
    }
#else
    { // fallback: stage A tile (already bf16) with straight 32B copies
      const int row = tid >> 1, ch = (tid & 1) * 16;
      const uint4* gp = (const uint4*)(Xb + (size_t)(m0 + row) * D_ + k0 + ch);
      uint4 u0 = gp[0], u1 = gp[1];
      uint4* lp = (uint4*)&sA[row * LDT + ch];
      lp[0] = u0; lp[1] = u1;
    }
#endif
    { // stage W tile fp32 -> bf16
      const int row = tid >> 2, ch = (tid & 3) * 8;
      const float4* gp = (const float4*)(W + (size_t)(n0 + row) * D_ + k0 + ch);
      float4 f0 = gp[0], f1 = gp[1];
      float t[8] = {f0.x, f0.y, f0.z, f0.w, f1.x, f1.y, f1.z, f1.w};
      union { __bf16 e[8]; uint4 q; } cv;
      #pragma unroll
      for (int i = 0; i < 8; i++) cv.e[i] = (__bf16)t[i];
      *(uint4*)&sB[row * LDT + ch] = cv.q;
      if (k0 + BK < D_)
        __builtin_prefetch(W + (size_t)(n0 + row) * D_ + k0 + BK + ch, 0, 1);
    }
    __syncthreads();

    FragBF a[2], bfr[2];
    #pragma unroll
    for (int i = 0; i < 2; i++) {
      const int r = wm * 32 + i * 16 + ln15;
      a[i].q[0] = *(const uint4*)&sA[r * LDT + hsel * 16];
      a[i].q[1] = *(const uint4*)&sA[r * LDT + hsel * 16 + 8];
    }
    #pragma unroll
    for (int j = 0; j < 2; j++) {
      const int c = wn * 32 + j * 16 + ln15;
      const uint4* p = (const uint4*)&sB[c * LDT + hsel * 16];
      bfr[j].q[0] = p[0]; bfr[j].q[1] = p[1];
    }
    #pragma unroll
    for (int i = 0; i < 2; i++)
      #pragma unroll
      for (int j = 0; j < 2; j++)
        acc[i][j].v = wmma_bf16(a[i].v, bfr[j].v, acc[i][j].v);
    __syncthreads();
  }

  #pragma unroll
  for (int j = 0; j < 2; j++) {
    const int n = n0 + wn * 32 + j * 16 + ln15;
    const float bval = bias[n];
    #pragma unroll
    for (int i = 0; i < 2; i++) {
      const int mbase = m0 + wm * 32 + i * 16 + (hsel << 3);
      #pragma unroll
      for (int r = 0; r < 8; r++)
        Out[(size_t)(mbase + r) * D_ + n] = acc[i][j].f[r] + bval;
    }
  }
}

// ---------------------------------------------------------------------------
extern "C" void kernel_launch(void* const* d_in, const int* in_sizes, int n_in,
                              void* d_out, int out_size, void* d_ws, size_t ws_size,
                              hipStream_t stream) {
  (void)in_sizes; (void)n_in; (void)out_size; (void)ws_size;
  const float*         query  = (const float*)d_in[0];
  const unsigned char* pmask  = (const unsigned char*)d_in[1];  // bool [B,S]
  const float*         qkv_w  = (const float*)d_in[2];
  const float*         qkv_b  = (const float*)d_in[3];
  const float*         out_w  = (const float*)d_in[4];
  const float*         out_b  = (const float*)d_in[5];
  float*               out    = (float*)d_out;

  const size_t NQ = (size_t)B_ * H_ * S_ * HD_;   // 4,194,304 elems
  __bf16* Qb = (__bf16*)d_ws;
  __bf16* Kb = Qb + NQ;
  __bf16* Vt = Kb + NQ;
  __bf16* Xb = Vt + NQ;                            // total 32 MB bf16 scratch

  qkv_gemm_kernel<<<32 * 48, 256, 0, stream>>>(query, qkv_w, qkv_b, Qb, Kb, Vt);
  attn_kernel<<<(B_ * H_ * (S_ / 16)) / 8, 256, 0, stream>>>(Qb, Kb, Vt, pmask, Xb);
  out_gemm_kernel<<<32 * 16, 256, 0, stream>>>(Xb, out_w, out_b, out);
}
